// MultiHeadAttention_15899968930535
// MI455X (gfx1250) — matrix operand order repaired and checked by reading, not compiled, not run on hardware
//
#include <hip/hip_runtime.h>

// =====================================================================
// MultiHeadAttention for MI455X (gfx1250), wave32 + v_wmma_f32_16x16x32_bf16.
// All GEMMs use split-bf16 (hi+lo) operands with fp32 accumulation:
//   a*b ~= a_hi*b_hi + a_hi*b_lo + a_lo*b_hi   (3 WMMAs per K-chunk)
// which recovers ~fp32 accuracy on the matrix cores.
//
// Addressing: all operand loads are (uniform SGPR base) + (loop-invariant
// 32-bit per-lane voffset) + (immediate K offset) so the backend can use
// saddr-form global_load_b128 with zero per-iteration address VALU,
// avoiding WMMA<->VALU hazard NOPs in the inner loop.
//
// Pipeline (B=4, N=2048, E=512, H=8, EH=4096, M=B*N=8192):
//   prep:   pack x -> hi/lo bf16; transpose+pack Wq/Wk/Wv and Wp
//   stage1: q,k = (B,H,E,N) packed; v stored transposed (B,H,N,E) packed
//   stage2: att_f32[b,h,e,f] = sum_n q*k          (NT GEMM, K=2048)
//   stage3: softmax rows of 512 (scale 1/sqrt(2048)), pack hi/lo
//   stage4: y[b,h,e,n] = sum_f att*v              (NT GEMM, K=512), packed
//   stage5: out[b,nn,c] = concat(y rows) @ Wp + bp (NT GEMM, K=4096, fp32)
// =====================================================================

typedef __bf16 bf16_t;
typedef bf16_t bf16x8  __attribute__((ext_vector_type(8)));
typedef bf16_t bf16x16 __attribute__((ext_vector_type(16)));
typedef float  f32x8   __attribute__((ext_vector_type(8)));

#define DEV static __device__ __forceinline__

DEV f32x8 wmma_bf16(bf16x16 a, bf16x16 b, f32x8 c) {
  // (neg_a, A, neg_b, B, c_mod, C, reuse_a, reuse_b)
  return __builtin_amdgcn_wmma_f32_16x16x32_bf16(false, a, false, b, (short)0, c,
                                                 false, false);
}

// A fragment (16x32 bf16, M = lane&15): off must already include row*ld + 8*hl.
// Lane holds K = 16h .. 16h+15 contiguous.
DEV bf16x16 load_a(const bf16_t* __restrict__ base, unsigned off) {
  bf16x8 u = *(const bf16x8*)(base + off + (off & 8u));
  bf16x8 v = *(const bf16x8*)(base + off + (off & 8u) + 8);
  return __builtin_shufflevector(u, v, 0, 1, 2, 3, 4, 5, 6, 7,
                                 8, 9, 10, 11, 12, 13, 14, 15);
}
// B fragment (32x16 bf16, Ncol = lane&15): off must include row*ld + 16*hl.
// Lane holds K = 16h .. 16h+15 contiguous.
DEV bf16x16 load_b(const bf16_t* __restrict__ base, unsigned off) {
  return *(const bf16x16*)(base + off);
}

DEV void split_f32(float v, bf16_t& h, bf16_t& l) {
  h = (bf16_t)v;
  l = (bf16_t)(v - (float)h);
}

// ---------------------------------------------------------------------
// prep kernels
// ---------------------------------------------------------------------
__global__ __launch_bounds__(256) void k_pack_split(const float* __restrict__ src,
                                                    bf16_t* __restrict__ hi,
                                                    bf16_t* __restrict__ lo, int n) {
  int i = blockIdx.x * blockDim.x + threadIdx.x;
  if (i < n) split_f32(src[i], hi[i], lo[i]);
}

// src is (R,C) row-major; writes transposed (C,R) hi/lo
__global__ __launch_bounds__(256) void k_transpose_split(const float* __restrict__ src,
                                                         bf16_t* __restrict__ hi,
                                                         bf16_t* __restrict__ lo,
                                                         int R, int C) {
  int i = blockIdx.x * blockDim.x + threadIdx.x;
  if (i < R * C) {
    int r = i / C, c = i % C;
    bf16_t h, l;
    split_f32(src[i], h, l);
    size_t o = (size_t)c * R + r;
    hi[o] = h;
    lo[o] = l;
  }
}

// ---------------------------------------------------------------------
// stage 1: projection.  C[j,i] = sum_p Wt[j,p]*x[i,p] + bias[j]
//   A = Wt (4096 x 512), B = x (8192 x 512).  Output scattered into
//   q/k (B,H,E,N) layout (vlayout=0) or v (B,H,N,E) layout (vlayout=1).
// ---------------------------------------------------------------------
__global__ __launch_bounds__(32) void k_proj(
    const bf16_t* __restrict__ wt_h, const bf16_t* __restrict__ wt_l,
    const bf16_t* __restrict__ x_h, const bf16_t* __restrict__ x_l,
    const float* __restrict__ bias, bf16_t* __restrict__ o_h,
    bf16_t* __restrict__ o_l, int vlayout) {
  const int lane = threadIdx.x, hl = lane >> 4, l16 = lane & 15;
  const int i0 = blockIdx.x * 64;  // token tile (N-dim of WMMA)
  const int j0 = blockIdx.y * 64;  // weight-column tile (M-dim of WMMA)

  unsigned arow[4], brow[4];
#pragma unroll
  for (int t = 0; t < 4; ++t) {
    arow[t] = (unsigned)(j0 + 16 * t + l16) * 512u + 8u * hl;
    brow[t] = (unsigned)(i0 + 16 * t + l16) * 512u + 16u * hl;
  }

  f32x8 acc[4][4] = {};
#pragma unroll 2
  for (int p = 0; p < 512; p += 32) {
    bf16x16 ah[4], al[4];
#pragma unroll
    for (int t = 0; t < 4; ++t) {
      ah[t] = load_a(wt_h, arow[t] + p);
      al[t] = load_a(wt_l, arow[t] + p);
    }
#pragma unroll
    for (int tb = 0; tb < 4; ++tb) {
      bf16x16 bh = load_b(x_h, brow[tb] + p);
      bf16x16 bl = load_b(x_l, brow[tb] + p);
#pragma unroll
      for (int ta = 0; ta < 4; ++ta) {
        acc[ta][tb] = wmma_bf16(ah[ta], bh, acc[ta][tb]);
        acc[ta][tb] = wmma_bf16(ah[ta], bl, acc[ta][tb]);
        acc[ta][tb] = wmma_bf16(al[ta], bh, acc[ta][tb]);
      }
    }
  }

#pragma unroll
  for (int ta = 0; ta < 4; ++ta) {
    int jbase = j0 + 16 * ta + 8 * hl;
    f32x8 bvec = *(const f32x8*)(bias + jbase);  // 32B aligned
#pragma unroll
    for (int tb = 0; tb < 4; ++tb) {
      int ii = i0 + 16 * tb + l16;  // token index
      int bb = ii >> 11, n = ii & 2047;
#pragma unroll
      for (int r = 0; r < 8; ++r) {
        int j = jbase + r;  // column of x@W, j = e*8 + h
        float v = acc[ta][tb][r] + bvec[r];
        int e = j >> 3, h = j & 7;
        bf16_t vh, vl;
        split_f32(v, vh, vl);
        size_t off = vlayout
                         ? ((size_t)((bb * 8 + h) * 2048 + n)) * 512 + e
                         : ((size_t)((bb * 8 + h) * 512 + e)) * 2048 + n;
        o_h[off] = vh;
        o_l[off] = vl;
      }
    }
  }
}

// ---------------------------------------------------------------------
// stage 2: att[e,f] = sum_n q[e,n]*k[f,n]   per z = b*8+h   (K = 2048)
// ---------------------------------------------------------------------
__global__ __launch_bounds__(32) void k_qk(
    const bf16_t* __restrict__ q_h, const bf16_t* __restrict__ q_l,
    const bf16_t* __restrict__ k_h, const bf16_t* __restrict__ k_l,
    float* __restrict__ att) {
  const int lane = threadIdx.x, hl = lane >> 4, l16 = lane & 15;
  const unsigned zb = (unsigned)blockIdx.z * 512u * 2048u;
  const int f0 = blockIdx.x * 64;  // N-dim (k rows)
  const int e0 = blockIdx.y * 64;  // M-dim (q rows)

  unsigned arow[4], brow[4];
#pragma unroll
  for (int t = 0; t < 4; ++t) {
    arow[t] = zb + (unsigned)(e0 + 16 * t + l16) * 2048u + 8u * hl;
    brow[t] = zb + (unsigned)(f0 + 16 * t + l16) * 2048u + 16u * hl;
  }

  f32x8 acc[4][4] = {};
#pragma unroll 2
  for (int p = 0; p < 2048; p += 32) {
    bf16x16 ah[4], al[4];
#pragma unroll
    for (int t = 0; t < 4; ++t) {
      ah[t] = load_a(q_h, arow[t] + p);
      al[t] = load_a(q_l, arow[t] + p);
    }
#pragma unroll
    for (int tb = 0; tb < 4; ++tb) {
      bf16x16 bh = load_b(k_h, brow[tb] + p);
      bf16x16 bl = load_b(k_l, brow[tb] + p);
#pragma unroll
      for (int ta = 0; ta < 4; ++ta) {
        acc[ta][tb] = wmma_bf16(ah[ta], bh, acc[ta][tb]);
        acc[ta][tb] = wmma_bf16(ah[ta], bl, acc[ta][tb]);
        acc[ta][tb] = wmma_bf16(al[ta], bh, acc[ta][tb]);
      }
    }
  }

  const size_t ab = (size_t)blockIdx.z * 512 * 512;
#pragma unroll
  for (int ta = 0; ta < 4; ++ta) {
    int ebase = e0 + 16 * ta + 8 * hl;
#pragma unroll
    for (int tb = 0; tb < 4; ++tb) {
      int f = f0 + 16 * tb + l16;
#pragma unroll
      for (int r = 0; r < 8; ++r)
        att[ab + (size_t)(ebase + r) * 512 + f] = acc[ta][tb][r];
    }
  }
}

// ---------------------------------------------------------------------
// stage 3: row softmax over 512 logits, scale = 1/sqrt(2048); pack hi/lo
// ---------------------------------------------------------------------
__global__ __launch_bounds__(256) void k_softmax(const float* __restrict__ att,
                                                 bf16_t* __restrict__ o_h,
                                                 bf16_t* __restrict__ o_l) {
  __shared__ float red[256];
  const int t = threadIdx.x;
  const float* p = att + (size_t)blockIdx.x * 512;
  const float scale = 0.022097086912079608f;  // 1/sqrt(2048)
  float v0 = p[t] * scale;
  float v1 = p[t + 256] * scale;
  red[t] = fmaxf(v0, v1);
  __syncthreads();
  for (int s = 128; s > 0; s >>= 1) {
    if (t < s) red[t] = fmaxf(red[t], red[t + s]);
    __syncthreads();
  }
  const float m = red[0];
  __syncthreads();
  float e0 = __expf(v0 - m), e1 = __expf(v1 - m);
  red[t] = e0 + e1;
  __syncthreads();
  for (int s = 128; s > 0; s >>= 1) {
    if (t < s) red[t] += red[t + s];
    __syncthreads();
  }
  const float inv = 1.0f / red[0];
  size_t o = (size_t)blockIdx.x * 512 + t;
  bf16_t h0, l0, h1, l1;
  split_f32(e0 * inv, h0, l0);
  split_f32(e1 * inv, h1, l1);
  o_h[o] = h0;        o_l[o] = l0;
  o_h[o + 256] = h1;  o_l[o + 256] = l1;
}

// ---------------------------------------------------------------------
// stage 4: y[e,n] = sum_f att[e,f]*vt[n,f]  per z   (K = 512)
//   att packed (z,512,512); vt packed (z,2048,512); y packed (z,512,2048)
// ---------------------------------------------------------------------
__global__ __launch_bounds__(32) void k_av(
    const bf16_t* __restrict__ a_h, const bf16_t* __restrict__ a_l,
    const bf16_t* __restrict__ vt_h, const bf16_t* __restrict__ vt_l,
    bf16_t* __restrict__ y_h, bf16_t* __restrict__ y_l) {
  const int lane = threadIdx.x, hl = lane >> 4, l16 = lane & 15;
  const unsigned ab = (unsigned)blockIdx.z * 512u * 512u;
  const unsigned vb = (unsigned)blockIdx.z * 2048u * 512u;
  const int n0 = blockIdx.x * 64;  // N-dim (vt rows)
  const int e0 = blockIdx.y * 64;  // M-dim (att rows)

  unsigned arow[4], brow[4];
#pragma unroll
  for (int t = 0; t < 4; ++t) {
    arow[t] = ab + (unsigned)(e0 + 16 * t + l16) * 512u + 8u * hl;
    brow[t] = vb + (unsigned)(n0 + 16 * t + l16) * 512u + 16u * hl;
  }

  f32x8 acc[4][4] = {};
#pragma unroll 2
  for (int p = 0; p < 512; p += 32) {
    bf16x16 ah[4], al[4];
#pragma unroll
    for (int t = 0; t < 4; ++t) {
      ah[t] = load_a(a_h, arow[t] + p);
      al[t] = load_a(a_l, arow[t] + p);
    }
#pragma unroll
    for (int tb = 0; tb < 4; ++tb) {
      bf16x16 bh = load_b(vt_h, brow[tb] + p);
      bf16x16 bl = load_b(vt_l, brow[tb] + p);
#pragma unroll
      for (int ta = 0; ta < 4; ++ta) {
        acc[ta][tb] = wmma_bf16(ah[ta], bh, acc[ta][tb]);
        acc[ta][tb] = wmma_bf16(ah[ta], bl, acc[ta][tb]);
        acc[ta][tb] = wmma_bf16(al[ta], bh, acc[ta][tb]);
      }
    }
  }

  const size_t yb = (size_t)blockIdx.z * 512 * 2048;
#pragma unroll
  for (int ta = 0; ta < 4; ++ta) {
    int ebase = e0 + 16 * ta + 8 * hl;
#pragma unroll
    for (int tb = 0; tb < 4; ++tb) {
      int n = n0 + 16 * tb + l16;
#pragma unroll
      for (int r = 0; r < 8; ++r) {
        bf16_t vh, vl;
        split_f32(acc[ta][tb][r], vh, vl);
        size_t off = yb + (size_t)(ebase + r) * 2048 + n;
        y_h[off] = vh;
        y_l[off] = vl;
      }
    }
  }
}

// ---------------------------------------------------------------------
// stage 5: out[ii, c] = sum_p A[ii,p]*WpT[c,p] + bp[c]   (K = 4096)
//   Row ii = b*2048+nn of the scrambled reshape is the concatenation of
//   y[b, 2*(nn&3), nn>>2, :] (p<2048) and y[b, 2*(nn&3)+1, nn>>2, :].
//   K split into two clean 2048 loops (one per segment).
// ---------------------------------------------------------------------
__global__ __launch_bounds__(32) void k_out(
    const bf16_t* __restrict__ y_h, const bf16_t* __restrict__ y_l,
    const bf16_t* __restrict__ wpt_h, const bf16_t* __restrict__ wpt_l,
    const float* __restrict__ bias, float* __restrict__ out) {
  const int lane = threadIdx.x, hl = lane >> 4, l16 = lane & 15;
  const int i0 = blockIdx.x * 64;  // token rows (M-dim)
  const int c0 = blockIdx.y * 64;  // output columns (N-dim)

  unsigned a0[4], a1[4], brow[4];
#pragma unroll
  for (int t = 0; t < 4; ++t) {
    int ii = i0 + 16 * t + l16;
    int bb = ii >> 11, nn = ii & 2047;
    int er = nn >> 2, pr = nn & 3;
    a0[t] = (unsigned)((bb * 8 + 2 * pr) * 512 + er) * 2048u + 8u * hl;
    a1[t] = (unsigned)((bb * 8 + 2 * pr + 1) * 512 + er) * 2048u + 8u * hl;
    brow[t] = (unsigned)(c0 + 16 * t + l16) * 4096u + 16u * hl;
  }

  f32x8 acc[4][4] = {};
#pragma unroll 2
  for (int p = 0; p < 2048; p += 32) {  // segment 0: heads 2*(nn&3)
    bf16x16 ah[4], al[4];
#pragma unroll
    for (int t = 0; t < 4; ++t) {
      ah[t] = load_a(y_h, a0[t] + p);
      al[t] = load_a(y_l, a0[t] + p);
    }
#pragma unroll
    for (int tb = 0; tb < 4; ++tb) {
      bf16x16 bh = load_b(wpt_h, brow[tb] + p);
      bf16x16 bl = load_b(wpt_l, brow[tb] + p);
#pragma unroll
      for (int ta = 0; ta < 4; ++ta) {
        acc[ta][tb] = wmma_bf16(ah[ta], bh, acc[ta][tb]);
        acc[ta][tb] = wmma_bf16(ah[ta], bl, acc[ta][tb]);
        acc[ta][tb] = wmma_bf16(al[ta], bh, acc[ta][tb]);
      }
    }
  }
#pragma unroll 2
  for (int p = 0; p < 2048; p += 32) {  // segment 1: heads 2*(nn&3)+1
    bf16x16 ah[4], al[4];
#pragma unroll
    for (int t = 0; t < 4; ++t) {
      ah[t] = load_a(y_h, a1[t] + p);
      al[t] = load_a(y_l, a1[t] + p);
    }
#pragma unroll
    for (int tb = 0; tb < 4; ++tb) {
      bf16x16 bh = load_b(wpt_h, brow[tb] + 2048 + p);
      bf16x16 bl = load_b(wpt_l, brow[tb] + 2048 + p);
#pragma unroll
      for (int ta = 0; ta < 4; ++ta) {
        acc[ta][tb] = wmma_bf16(ah[ta], bh, acc[ta][tb]);
        acc[ta][tb] = wmma_bf16(ah[ta], bl, acc[ta][tb]);
        acc[ta][tb] = wmma_bf16(al[ta], bh, acc[ta][tb]);
      }
    }
  }

#pragma unroll
  for (int ta = 0; ta < 4; ++ta) {
    int rbase = i0 + 16 * ta + 8 * hl;
#pragma unroll
    for (int tb = 0; tb < 4; ++tb) {
      int c = c0 + 16 * tb + l16;
      float bv = bias[c];
#pragma unroll
      for (int r = 0; r < 8; ++r) {
        size_t ii = (size_t)(rbase + r);
        out[ii * 512 + c] = acc[ta][tb][r] + bv;
      }
    }
  }
}

// ---------------------------------------------------------------------
// host launcher
// ---------------------------------------------------------------------
extern "C" void kernel_launch(void* const* d_in, const int* in_sizes, int n_in,
                              void* d_out, int out_size, void* d_ws, size_t ws_size,
                              hipStream_t stream) {
  (void)in_sizes; (void)n_in; (void)out_size; (void)ws_size;
  const float* x  = (const float*)d_in[0];
  const float* Wq = (const float*)d_in[1];
  const float* bq = (const float*)d_in[2];
  const float* Wk = (const float*)d_in[3];
  const float* bk = (const float*)d_in[4];
  const float* Wv = (const float*)d_in[5];
  const float* bv = (const float*)d_in[6];
  const float* Wp = (const float*)d_in[7];
  const float* bp = (const float*)d_in[8];
  float* out = (float*)d_out;

  const int E = 512, H = 8, Bn = 4, Nn = 2048;
  const int EH = E * H;                        // 4096
  const int M = Bn * Nn;                       // 8192
  const size_t QKV = (size_t)Bn * H * E * Nn;  // 33,554,432 elems
  const size_t ATT = (size_t)Bn * H * E * E;   //  8,388,608 elems

  char* w = (char*)d_ws;
  auto take = [&](size_t bytes) -> char* {
    char* p = w;
    w += (bytes + 255) & ~(size_t)255;
    return p;
  };

  bf16_t* x_h = (bf16_t*)take((size_t)M * E * 2);
  bf16_t* x_l = (bf16_t*)take((size_t)M * E * 2);
  bf16_t* wqt_h = (bf16_t*)take((size_t)E * EH * 2);
  bf16_t* wqt_l = (bf16_t*)take((size_t)E * EH * 2);
  bf16_t* wkt_h = (bf16_t*)take((size_t)E * EH * 2);
  bf16_t* wkt_l = (bf16_t*)take((size_t)E * EH * 2);
  bf16_t* wvt_h = (bf16_t*)take((size_t)E * EH * 2);
  bf16_t* wvt_l = (bf16_t*)take((size_t)E * EH * 2);
  bf16_t* wpt_h = (bf16_t*)take((size_t)EH * E * 2);
  bf16_t* wpt_l = (bf16_t*)take((size_t)EH * E * 2);
  bf16_t* q_h = (bf16_t*)take(QKV * 2);
  bf16_t* q_l = (bf16_t*)take(QKV * 2);
  bf16_t* kk_h = (bf16_t*)take(QKV * 2);
  bf16_t* kk_l = (bf16_t*)take(QKV * 2);
  bf16_t* vt_h = (bf16_t*)take(QKV * 2);
  bf16_t* vt_l = (bf16_t*)take(QKV * 2);
  float* attf = (float*)take(ATT * 4);
  bf16_t* at_h = (bf16_t*)take(ATT * 2);
  bf16_t* at_l = (bf16_t*)take(ATT * 2);
  bf16_t* y_h = (bf16_t*)take(QKV * 2);
  bf16_t* y_l = (bf16_t*)take(QKV * 2);

  // prep
  k_pack_split<<<(M * E) / 256, 256, 0, stream>>>(x, x_h, x_l, M * E);
  k_transpose_split<<<(E * EH) / 256, 256, 0, stream>>>(Wq, wqt_h, wqt_l, E, EH);
  k_transpose_split<<<(E * EH) / 256, 256, 0, stream>>>(Wk, wkt_h, wkt_l, E, EH);
  k_transpose_split<<<(E * EH) / 256, 256, 0, stream>>>(Wv, wvt_h, wvt_l, E, EH);
  k_transpose_split<<<(EH * E) / 256, 256, 0, stream>>>(Wp, wpt_h, wpt_l, EH, E);

  // stage 1: projections
  dim3 gp(M / 64, EH / 64);
  k_proj<<<gp, 32, 0, stream>>>(wqt_h, wqt_l, x_h, x_l, bq, q_h, q_l, 0);
  k_proj<<<gp, 32, 0, stream>>>(wkt_h, wkt_l, x_h, x_l, bk, kk_h, kk_l, 0);
  k_proj<<<gp, 32, 0, stream>>>(wvt_h, wvt_l, x_h, x_l, bv, vt_h, vt_l, 1);

  // stage 2: logits
  dim3 gq(E / 64, E / 64, Bn * H);
  k_qk<<<gq, 32, 0, stream>>>(q_h, q_l, kk_h, kk_l, attf);

  // stage 3: softmax
  k_softmax<<<Bn * H * E, 256, 0, stream>>>(attf, at_h, at_l);

  // stage 4: att @ V
  dim3 ga(Nn / 64, E / 64, Bn * H);
  k_av<<<ga, 32, 0, stream>>>(at_h, at_l, vt_h, vt_l, y_h, y_l);

  // stage 5: output projection (reshape-scramble folded into row gather)
  dim3 go(M / 64, E / 64);
  k_out<<<go, 32, 0, stream>>>(y_h, y_l, wpt_h, wpt_l, bp, out);
}
